// QuadraticConv2D_73194832659172
// MI455X (gfx1250) — matrix operand order repaired and checked by reading, not compiled, not run on hardware
//
#include <hip/hip_runtime.h>

typedef _Float16 h16;
typedef __attribute__((ext_vector_type(8)))  _Float16 v8h;
typedef __attribute__((ext_vector_type(16))) _Float16 v16h;
typedef __attribute__((ext_vector_type(8)))  float    v8f;

// pair tables for the 55 (i<=j) quadratic terms over 10 S-terms (compile-time only)
constexpr unsigned char cII[55] = {
  0,0,0,0,0,0,0,0,0,0,
  1,1,1,1,1,1,1,1,1,
  2,2,2,2,2,2,2,2,
  3,3,3,3,3,3,3,
  4,4,4,4,4,4,
  5,5,5,5,5,
  6,6,6,6,
  7,7,7,
  8,8,
  9};
constexpr unsigned char cJJ[55] = {
  0,1,2,3,4,5,6,7,8,9,
  1,2,3,4,5,6,7,8,9,
  2,3,4,5,6,7,8,9,
  3,4,5,6,7,8,9,
  4,5,6,7,8,9,
  5,6,7,8,9,
  6,7,8,9,
  7,8,9,
  8,9,
  9};
// LDS-halo offset (in halves) for S-term t>=1: ((t-1)/3 * 10 + (t-1)%3) * 64
constexpr int cTOff[10] = {0, 0, 64, 128, 640, 704, 768, 1280, 1344, 1408};

__global__ void prep_weights_f16(const float* __restrict__ k, h16* __restrict__ w, int n) {
  int i = blockIdx.x * 256 + threadIdx.x;
  if (i < n) w[i] = (h16)k[i];
}

// Load the 4 B fragments (column N=lane%16, K in A-fragment order) of K-tile kt.
// After full unroll, kt*4096 is a literal folded into the 24-bit global offset.
__device__ __forceinline__ void load_b_tile(const h16* __restrict__ bbase, int kt, v16h B[4]) {
  const h16* br = bbase + (size_t)kt * 4096;   // 32 K-rows * 128 cols per tile
  #pragma unroll
  for (int q = 0; q < 4; ++q) {
    #pragma unroll
    for (int s = 0; s < 16; ++s) B[q][s] = br[(s < 8 ? s : s + 8) * 128 + q * 16];
  }
}

// One workgroup = 8x8 pixel tile of one image, all 128 outputs.
// 256 threads = 8 waves (wave32), arranged 4(M) x 2(N).
__global__ __launch_bounds__(256) void quadconv2d_wmma(
    const float* __restrict__ x,    // [8,56,56,64] fp32
    const h16*   __restrict__ Bw,   // [3520,128] fp16 (pair*64+ch rows)
    float*       __restrict__ out)  // [8,56,56,128] fp32
{
  __shared__ __align__(16) h16 sHalo[10 * 10 * 64]; // input halo, fp16

  const int tid = threadIdx.x;
  const int b  = blockIdx.z;
  const int h0 = blockIdx.y * 8;
  const int w0 = blockIdx.x * 8;

  // ---- stage 10x10x64 halo (zero-padded), fp32 -> fp16 ----
  for (int u = tid; u < 6400; u += 256) {
    int c = u & 63;
    int rest = u >> 6;            // y*10 + x
    int xq = rest % 10, yq = rest / 10;
    int gh = h0 - 1 + yq, gw = w0 - 1 + xq;
    float v = 0.0f;
    if ((unsigned)gh < 56u && (unsigned)gw < 56u)
      v = x[(((size_t)b * 56 + gh) * 56 + gw) * 64 + c];
    sHalo[u] = (h16)v;
  }
  __syncthreads();

  const int lane  = tid & 31;
  const int wid   = tid >> 5;
  const int waveM = wid & 3;      // pixel-row tile (16 rows each)
  const int waveN = wid >> 2;     // output-col tile (64 cols each)
  const int laneM = lane & 15;
  const int mA    = waveM * 16 + laneM;              // pixel row this lane feeds into A
  const int haloLane = ((mA >> 3) * 10 + (mA & 7)) * 64;
  const int klo   = (lane < 16) ? 0 : 8;             // 16-bit A-fragment layout
  const int n0    = waveN * 64;

  // per-lane bases; all per-tile addressing folds into immediate offsets
  const h16* hbase = sHalo + haloLane + klo;
  const h16* bbase = Bw + (size_t)klo * 128 + n0 + laneM;   // column N=laneM, K offset klo

  const v8h vone = {(h16)1.f,(h16)1.f,(h16)1.f,(h16)1.f,
                    (h16)1.f,(h16)1.f,(h16)1.f,(h16)1.f};

  v8f acc0 = {}, acc1 = {}, acc2 = {}, acc3 = {};

  // software pipeline: B fragments prefetched one K-tile ahead
  v16h Bcur[4], Bnxt[4];
  load_b_tile(bbase, 0, Bcur);

  // K = 55 pairs * 64 channels, fully unrolled: 110 K-tiles of 32
  #pragma unroll
  for (int kt = 0; kt < 110; ++kt) {
    if (kt + 1 < 110) load_b_tile(bbase, kt + 1, Bnxt);   // issue next clause early

    const int p  = kt >> 1;
    const int c0 = (kt & 1) << 5;
    const int ii = cII[p];
    const int jj = cJJ[p];

    // A fragment: lane<16 -> K {0..7,16..23}; lane>=16 -> K {8..15,24..31}
    v8h A0, A1;
    if (ii == 0 && jj == 0) {            // bias pair: ones
      A0 = vone; A1 = vone;
    } else if (ii == 0) {                // linear pair: A = S_j
      A0 = *(const v8h*)(hbase + cTOff[jj] + c0);
      A1 = *(const v8h*)(hbase + cTOff[jj] + c0 + 16);
    } else if (ii == jj) {               // diagonal: square one load
      v8h s0 = *(const v8h*)(hbase + cTOff[ii] + c0);
      v8h s1 = *(const v8h*)(hbase + cTOff[ii] + c0 + 16);
      A0 = s0 * s0; A1 = s1 * s1;
    } else {                             // general product
      v8h a0 = *(const v8h*)(hbase + cTOff[ii] + c0);
      v8h a1 = *(const v8h*)(hbase + cTOff[ii] + c0 + 16);
      v8h s0 = *(const v8h*)(hbase + cTOff[jj] + c0);
      v8h s1 = *(const v8h*)(hbase + cTOff[jj] + c0 + 16);
      A0 = a0 * s0; A1 = a1 * s1;        // v_pk_mul_f16
    }
    union { v16h v; v8h h[2]; } A; A.h[0] = A0; A.h[1] = A1;

    acc0 = __builtin_amdgcn_wmma_f32_16x16x32_f16(false, A.v, false, Bcur[0], (short)0, acc0, false, false);
    acc1 = __builtin_amdgcn_wmma_f32_16x16x32_f16(false, A.v, false, Bcur[1], (short)0, acc1, false, false);
    acc2 = __builtin_amdgcn_wmma_f32_16x16x32_f16(false, A.v, false, Bcur[2], (short)0, acc2, false, false);
    acc3 = __builtin_amdgcn_wmma_f32_16x16x32_f16(false, A.v, false, Bcur[3], (short)0, acc3, false, false);

    #pragma unroll
    for (int q = 0; q < 4; ++q) Bcur[q] = Bnxt[q];  // register rename, no copies
  }

  // ---- store D: lane<16: (M=g, N=lane); lane>=16: (M=8+g, N=lane-16) ----
  const int colBase = n0 + laneM;
  const int rowAdd  = (lane < 16) ? 0 : 8;
  #pragma unroll
  for (int g = 0; g < 8; ++g) {
    int mrow = waveM * 16 + g + rowAdd;
    int hh = mrow >> 3, ww = mrow & 7;
    size_t base = (((size_t)b * 56 + (h0 + hh)) * 56 + (w0 + ww)) * 128;
    out[base + colBase +  0] = acc0[g];
    out[base + colBase + 16] = acc1[g];
    out[base + colBase + 32] = acc2[g];
    out[base + colBase + 48] = acc3[g];
  }
}

extern "C" void kernel_launch(void* const* d_in, const int* in_sizes, int n_in,
                              void* d_out, int out_size, void* d_ws, size_t ws_size,
                              hipStream_t stream) {
  const float* x = (const float*)d_in[0];   // inputs [8,56,56,64]
  const float* k = (const float*)d_in[1];   // kernel [55,64,128]
  h16* wf16 = (h16*)d_ws;                   // 3520*128*2 = 901120 bytes
  const int nW = 55 * 64 * 128;
  prep_weights_f16<<<(nW + 255) / 256, 256, 0, stream>>>(k, wf16, nW);

  dim3 grid(7, 7, 8);                       // 7x7 tiles of 8x8 pixels, 8 images
  quadconv2d_wmma<<<grid, 256, 0, stream>>>(x, wf16, (float*)d_out);
}
